// Main_Model_32727650796224
// MI455X (gfx1250) — matrix operand order repaired and checked by reading, not compiled, not run on hardware
//
#include <hip/hip_runtime.h>

// ---------------------------------------------------------------------------
// Fused draft-prediction model for MI455X (gfx1250, wave32, WMMA).
// One wave owns 16 samples end-to-end; all intermediates stay in LDS.
// All GEMMs run as V_WMMA_F32_16X16X32_F16 (f16 in, f32 accumulate).
// Hot-loop weight fragments are pinned in VGPRs (loop-invariant B matrices).
// ---------------------------------------------------------------------------

typedef __attribute__((ext_vector_type(16))) _Float16 half16;
typedef __attribute__((ext_vector_type(8)))  _Float16 half8;
typedef __attribute__((ext_vector_type(4)))  _Float16 half4;
typedef __attribute__((ext_vector_type(8)))  float    float8;

// ---- LDS weight arena offsets (in halves). Weights stored transposed:
// wt[n * Kpad + k], K padded to mult of 32 and zero-filled, N padded to 16.
constexpr int WT_H1 = 0;              // 32 x 160
constexpr int WT_H2 = WT_H1 + 32*160; // 32 x 32
constexpr int WT_H3 = WT_H2 + 32*32;  // 16 x 32
constexpr int WT_C1 = WT_H3 + 16*32;  // 16 x 32
constexpr int WT_C2 = WT_C1 + 16*32;  // 16 x 32 (K=16 padded)
constexpr int WT_C3 = WT_C2 + 16*32;  // 16 x 32 (N=8 padded, K=16 padded)
constexpr int WT_T1 = WT_C3 + 16*32;  // 64 x 96
constexpr int WT_T2 = WT_T1 + 64*96;  // 64 x 64
constexpr int WT_T3 = WT_T2 + 64*64;  // 16 x 64
constexpr int WT_F1 = WT_T3 + 16*64;  // 16 x 32
constexpr int WT_F2 = WT_F1 + 16*32;  // 16 x 32 (K=16 padded)
constexpr int WT_F3 = WT_F2 + 16*32;  // 16 x 32 (N=2, K=16 padded)
constexpr int WT_TOTAL = WT_F3 + 16*32; // 20992 halves = 41984 B

// ---- bias arena offsets (floats), padded with zeros
constexpr int B_H1 = 0,  B_H2 = 32,  B_H3 = 64,  B_C1 = 80,  B_C2 = 96,  B_C3 = 112;
constexpr int B_T1 = 128, B_T2 = 192, B_T3 = 256, B_F1 = 272, B_F2 = 288, B_F3 = 304;
constexpr int B_TOTAL = 320;

// ---- per-wave arena layout (halves); scratch region is phase-overlapped
constexpr int A_HO   = 0;     // 16 samples x 10 heroes x 16 ch      (2560)
constexpr int A_TIN  = 2560;  // 32 rows (team-major) x 96           (3072)
constexpr int A_TOUT = 5632;  // 32 rows x 16                        (512)
constexpr int A_SCR  = 6144;  // phase scratch                       (4096)
constexpr int A_SIZE = 10240; // 20480 B per wave

// ---------------------------------------------------------------------------
// Fragment helpers (layouts per CDNA5 ISA 7.12.2, wave32)
// ---------------------------------------------------------------------------
__device__ __forceinline__ half16 mk16(half8 lo, half8 hi) {
  return __builtin_shufflevector(lo, hi, 0,1,2,3,4,5,6,7,8,9,10,11,12,13,14,15);
}
__device__ __forceinline__ half8 lds8(const _Float16* p) { return *(const half8*)p; }

// A: 16x32 f16, row-major tile in LDS. lane(0-15): rows, halves K=kb..kb+7 and
// K=16+kb..16+kb+7 where kb = (lane>=16)*8.
__device__ __forceinline__ half16 fragA(const _Float16* base, int stride, int k0, int lane) {
  int r = lane & 15, kb = (lane >> 4) * 8;
  const _Float16* p = base + r * stride + k0 + kb;
  return mk16(lds8(p), lds8(p + 16));
}
// A with logical K=16 (upper 16 K-values are exact zeros)
__device__ __forceinline__ half16 fragA_k16(const _Float16* base, int stride, int lane) {
  int r = lane & 15, kb = (lane >> 4) * 8;
  half8 z = {};
  return mk16(lds8(base + r * stride + kb), z);
}
// A assembled from two 8-half runs at arbitrary LDS addresses (pair/concat rows)
__device__ __forceinline__ half16 fragA2(const _Float16* p0, const _Float16* p1) {
  return mk16(lds8(p0), lds8(p1));
}
// B: 32x16 f16 from transposed weights wt[n*Kpad + k].
// lane: col = lane%16, K=kb..kb+7 and K=16+kb..16+kb+7 where kb = (lane>=16)*8.
__device__ __forceinline__ half16 fragB(const _Float16* wt, int Kpad, int ncol0, int k0, int lane) {
  int c = lane & 15, kb = (lane >> 4) * 8;
  const _Float16* p = wt + (ncol0 + c) * Kpad + k0 + kb;
  return mk16(lds8(p), lds8(p + 16));
}
__device__ __forceinline__ float8 wmma16(half16 a, half16 b, float8 c) {
  // D = A * B + C ; emits v_wmma_f32_16x16x32_f16
  return __builtin_amdgcn_wmma_f32_16x16x32_f16(false, a, false, b, (short)0, c, false, false);
}
// D (f32 16x16: row = e + 8*(lane>=16), col = lane%16) -> f16 LDS tile,
// with bias (per output channel = col) and optional ReLU.
__device__ __forceinline__ void storeD(_Float16* dst, int stride, float8 acc,
                                       const float* bias, int ncol0, bool relu, int lane) {
  int c = lane & 15, r0 = (lane >> 4) * 8;
  float bv = bias[ncol0 + c];
#pragma unroll
  for (int e = 0; e < 8; ++e) {
    float v = acc[e] + bv;
    if (relu) v = fmaxf(v, 0.0f);
    dst[(r0 + e) * stride + c] = (_Float16)v;
  }
}

// ---------------------------------------------------------------------------
// Weight / bias staging: fp32 global -> f16 transposed LDS, zero padded
// ---------------------------------------------------------------------------
__device__ __forceinline__ void stageW(_Float16* dst, const float* src,
                                       int K, int Kpad, int N, int Npad,
                                       int tid, int nthr) {
  for (int idx = tid; idx < Npad * Kpad; idx += nthr) {
    int n = idx / Kpad, k = idx - n * Kpad;
    dst[idx] = (n < N && k < K) ? (_Float16)src[k * N + n] : (_Float16)0.0f;
  }
}
__device__ __forceinline__ void stageBias(float* dst, const float* src,
                                          int N, int Npad, int tid, int nthr) {
  for (int idx = tid; idx < Npad; idx += nthr) dst[idx] = (idx < N) ? src[idx] : 0.0f;
}

// ---------------------------------------------------------------------------
__global__ __launch_bounds__(128, 1)
void draft_fused_kernel(const float* __restrict__ x,
                        const float* hw1, const float* hb1,
                        const float* hw2, const float* hb2,
                        const float* hw3, const float* hb3,
                        const float* cw1, const float* cb1,
                        const float* cw2, const float* cb2,
                        const float* cw3, const float* cb3,
                        const float* tw1, const float* tb1,
                        const float* tw2, const float* tb2,
                        const float* tw3, const float* tb3,
                        const float* fw1, const float* fb1,
                        const float* fw2, const float* fb2,
                        const float* fw3, const float* fb3,
                        float* __restrict__ out, int nSamples) {
  __shared__ __align__(16) _Float16 s_w[WT_TOTAL];
  __shared__ float s_b[B_TOTAL];
  __shared__ __align__(16) _Float16 s_mm[4][A_SIZE];

  const int tid  = threadIdx.x;
  const int nthr = blockDim.x;
  const int lane = tid & 31;
  const int wave = tid >> 5;

  // ---- stage all weights (f16, transposed, padded) + biases into LDS
  stageW(s_w + WT_H1, hw1, 140, 160, 32, 32, tid, nthr);
  stageW(s_w + WT_H2, hw2,  32,  32, 32, 32, tid, nthr);
  stageW(s_w + WT_H3, hw3,  32,  32, 16, 16, tid, nthr);
  stageW(s_w + WT_C1, cw1,  32,  32, 16, 16, tid, nthr);
  stageW(s_w + WT_C2, cw2,  16,  32, 16, 16, tid, nthr);
  stageW(s_w + WT_C3, cw3,  16,  32,  8, 16, tid, nthr);
  stageW(s_w + WT_T1, tw1,  84,  96, 64, 64, tid, nthr);
  stageW(s_w + WT_T2, tw2,  64,  64, 64, 64, tid, nthr);
  stageW(s_w + WT_T3, tw3,  64,  64, 16, 16, tid, nthr);
  stageW(s_w + WT_F1, fw1,  32,  32, 16, 16, tid, nthr);
  stageW(s_w + WT_F2, fw2,  16,  32, 16, 16, tid, nthr);
  stageW(s_w + WT_F3, fw3,  16,  32,  2, 16, tid, nthr);
  stageBias(s_b + B_H1, hb1, 32, 32, tid, nthr);
  stageBias(s_b + B_H2, hb2, 32, 32, tid, nthr);
  stageBias(s_b + B_H3, hb3, 16, 16, tid, nthr);
  stageBias(s_b + B_C1, cb1, 16, 16, tid, nthr);
  stageBias(s_b + B_C2, cb2, 16, 16, tid, nthr);
  stageBias(s_b + B_C3, cb3,  8, 16, tid, nthr);
  stageBias(s_b + B_T1, tb1, 64, 64, tid, nthr);
  stageBias(s_b + B_T2, tb2, 64, 64, tid, nthr);
  stageBias(s_b + B_T3, tb3, 16, 16, tid, nthr);
  stageBias(s_b + B_F1, fb1, 16, 16, tid, nthr);
  stageBias(s_b + B_F2, fb2, 16, 16, tid, nthr);
  stageBias(s_b + B_F3, fb3,  2, 16, tid, nthr);
  __syncthreads();

  const int sampleBase = (blockIdx.x * 4 + wave) * 16;
  if (sampleBase >= nSamples) return;

  _Float16* arena = s_mm[wave];
  _Float16* ho   = arena + A_HO;    // [16 samples][10 heroes][16]
  _Float16* tin  = arena + A_TIN;   // [2 teams * 16 samples][96]
  _Float16* tout = arena + A_TOUT;  // [2 teams * 16 samples][16]
  _Float16* scr  = arena + A_SCR;

  const float8 zf = {};

  // =====================================================================
  // Stage 1: hero MLP 140->32->32->16 over 160 rows (10 M-tiles of 16).
  // All 13 weight fragments are loop-invariant -> pin them in VGPRs so the
  // hot loop only streams A-tiles through the DS pipe.
  // =====================================================================
  {
    _Float16* xt  = scr;            // [16][160] f16 staged input tile
    _Float16* h1o = scr + 2560;     // [16][32]
    _Float16* h2o = scr + 3072;     // [16][32]
    const long rowBase = (long)sampleBase * 10;

    half16 hB1[10];
#pragma unroll
    for (int ks = 0; ks < 5; ++ks) {
      hB1[ks * 2 + 0] = fragB(s_w + WT_H1, 160, 0,  ks * 32, lane);
      hB1[ks * 2 + 1] = fragB(s_w + WT_H1, 160, 16, ks * 32, lane);
    }
    half16 hB2a = fragB(s_w + WT_H2, 32, 0,  0, lane);
    half16 hB2b = fragB(s_w + WT_H2, 32, 16, 0, lane);
    half16 hB3  = fragB(s_w + WT_H3, 32, 0,  0, lane);

    for (int t = 0; t < 10; ++t) {
      // coalesced float4 loads: 16 rows x 140 f32 -> f16 LDS, pad K to 160
      const float4* xg4 = (const float4*)(x + (size_t)(rowBase + 16 * t) * 140);
      for (int idx = lane; idx < 560; idx += 32) {
        int r = idx / 35, c4 = idx - r * 35;
        float4 v = xg4[r * 35 + c4];
        half4 h = {(_Float16)v.x, (_Float16)v.y, (_Float16)v.z, (_Float16)v.w};
        *(half4*)(xt + r * 160 + c4 * 4) = h;
      }
      for (int idx = lane; idx < 16 * 20; idx += 32) {
        int r = idx / 20, cc = idx - r * 20;
        xt[r * 160 + 140 + cc] = (_Float16)0.0f;
      }
      if (t < 9) __builtin_prefetch((const char*)(xg4 + 560) + lane * 128, 0, 1);
      __builtin_amdgcn_wave_barrier();

      // L1: K=160 (5 ksteps) x N=32 (2 ntiles), weights from registers
      float8 a0 = zf, a1 = zf;
#pragma unroll
      for (int ks = 0; ks < 5; ++ks) {
        half16 A = fragA(xt, 160, ks * 32, lane);
        a0 = wmma16(A, hB1[ks * 2 + 0], a0);
        a1 = wmma16(A, hB1[ks * 2 + 1], a1);
      }
      storeD(h1o, 32, a0, s_b + B_H1, 0,  true, lane);
      storeD(h1o, 32, a1, s_b + B_H1, 16, true, lane);
      __builtin_amdgcn_wave_barrier();

      // L2: K=32, N=32
      {
        half16 A = fragA(h1o, 32, 0, lane);
        float8 b0 = wmma16(A, hB2a, zf);
        float8 b1 = wmma16(A, hB2b, zf);
        storeD(h2o, 32, b0, s_b + B_H2, 0,  true, lane);
        storeD(h2o, 32, b1, s_b + B_H2, 16, true, lane);
      }
      __builtin_amdgcn_wave_barrier();

      // L3: K=32, N=16 (linear) -> scatter into ho[s][h][16]
      {
        half16 A = fragA(h2o, 32, 0, lane);
        float8 d3 = wmma16(A, hB3, zf);
        int c = lane & 15, r0 = (lane >> 4) * 8;
        float bv = s_b[B_H3 + c];
#pragma unroll
        for (int e = 0; e < 8; ++e) {
          int wr = 16 * t + r0 + e;       // wave-local hero row
          int s = wr / 10, h = wr - s * 10;
          ho[s * 160 + h * 16 + c] = (_Float16)(d3[e] + bv);
        }
      }
      __builtin_amdgcn_wave_barrier();
    }
  }

  // =====================================================================
  // Stage 2: counter MLP 32->16->16->8 over 50 pairs; fused max-reduce.
  // The 3 weight fragments are pinned in registers across the pair loop.
  // (samples live in the M dimension, pairs span loop iterations -> elementwise D-fragment max)
  // =====================================================================
  float8 g1, g2;
#pragma unroll
  for (int e = 0; e < 8; ++e) { g1[e] = -3.0e38f; g2[e] = -3.0e38f; }
  {
    _Float16* c1o = scr;        // [16][16]
    _Float16* c2o = scr + 256;  // [16][16]
    const int r = lane & 15, kb = (lane >> 4) * 8;
    const half16 cB1 = fragB(s_w + WT_C1, 32, 0, 0, lane);
    const half16 cB2 = fragB(s_w + WT_C2, 32, 0, 0, lane);
    const half16 cB3 = fragB(s_w + WT_C3, 32, 0, 0, lane);
    const float  bv3 = s_b[B_C3 + (lane & 15)];
    for (int p = 0; p < 50; ++p) {
      int i, j;
      if (p < 25) { i = p / 5;            j = 5 + (p - (p / 5) * 5); }
      else        { int q = p - 25; i = 5 + q / 5; j = q - (q / 5) * 5; }
      // pair input row = concat(ho[s][i], ho[s][j]) built directly as A-frag
      half16 A = fragA2(ho + r * 160 + i * 16 + kb,
                        ho + r * 160 + j * 16 + kb);
      float8 d = wmma16(A, cB1, zf);
      storeD(c1o, 16, d, s_b + B_C1, 0, true, lane);
      __builtin_amdgcn_wave_barrier();
      d = wmma16(fragA_k16(c1o, 16, lane), cB2, zf);
      storeD(c2o, 16, d, s_b + B_C2, 0, true, lane);
      __builtin_amdgcn_wave_barrier();
      d = wmma16(fragA_k16(c2o, 16, lane), cB3, zf);
#pragma unroll
      for (int e = 0; e < 8; ++e) {
        float v = d[e] + bv3;
        if (p < 25) g1[e] = fmaxf(g1[e], v);
        else        g2[e] = fmaxf(g2[e], v);
      }
      __builtin_amdgcn_wave_barrier();
    }
  }

  // =====================================================================
  // Stage 3: channel pool (pairs of channels via shfl_xor) + team input
  //          tin rows: team*16 + sample ; K = 80 hero + 4 pooled + 12 pad
  // =====================================================================
  {
    for (int idx = lane; idx < 32 * 80; idx += 32) {   // hero features
      int tr = idx / 80, k = idx - tr * 80;
      int tm = tr >> 4, s = tr & 15;
      tin[tr * 96 + k] = ho[s * 160 + tm * 80 + k];
    }
    for (int idx = lane; idx < 32 * 12; idx += 32) {   // K pad 84..95
      int tr = idx / 12, k = idx - tr * 12;
      tin[tr * 96 + 84 + k] = (_Float16)0.0f;
    }
    float8 q1, q2;
#pragma unroll
    for (int e = 0; e < 8; ++e) {
      q1[e] = fmaxf(g1[e], __shfl_xor(g1[e], 1, 32));
      q2[e] = fmaxf(g2[e], __shfl_xor(g2[e], 1, 32));
    }
    int c = lane & 15, r0 = (lane >> 4) * 8;
    if ((c & 1) == 0 && c < 8) {
      int ch = c >> 1;
#pragma unroll
      for (int e = 0; e < 8; ++e) {
        int s = r0 + e;
        tin[(0  + s) * 96 + 80 + ch] = (_Float16)q1[e];
        tin[(16 + s) * 96 + 80 + ch] = (_Float16)q2[e];
      }
    }
    __builtin_amdgcn_wave_barrier();
  }

  // =====================================================================
  // Stage 4: team MLP 84->64->64->16 over 32 rows (2 M-tiles).
  // Runs once per wave; weight fragments loaded inline (22 frags would
  // overflow reasonable register budget, and this stage is cold).
  // =====================================================================
  {
    _Float16* t1o = scr;         // [32][64]
    _Float16* t2o = scr + 2048;  // [32][64]
#pragma unroll
    for (int mt = 0; mt < 2; ++mt) {
      const _Float16* ain = tin + mt * 16 * 96;
      float8 acc[4] = {zf, zf, zf, zf};
#pragma unroll
      for (int ks = 0; ks < 3; ++ks) {
        half16 A = fragA(ain, 96, ks * 32, lane);
#pragma unroll
        for (int nt = 0; nt < 4; ++nt)
          acc[nt] = wmma16(A, fragB(s_w + WT_T1, 96, nt * 16, ks * 32, lane), acc[nt]);
      }
#pragma unroll
      for (int nt = 0; nt < 4; ++nt)
        storeD(t1o + mt * 16 * 64, 64, acc[nt], s_b + B_T1, nt * 16, true, lane);
      __builtin_amdgcn_wave_barrier();

      float8 acc2[4] = {zf, zf, zf, zf};
#pragma unroll
      for (int ks = 0; ks < 2; ++ks) {
        half16 A = fragA(t1o + mt * 16 * 64, 64, ks * 32, lane);
#pragma unroll
        for (int nt = 0; nt < 4; ++nt)
          acc2[nt] = wmma16(A, fragB(s_w + WT_T2, 64, nt * 16, ks * 32, lane), acc2[nt]);
      }
#pragma unroll
      for (int nt = 0; nt < 4; ++nt)
        storeD(t2o + mt * 16 * 64, 64, acc2[nt], s_b + B_T2, nt * 16, true, lane);
      __builtin_amdgcn_wave_barrier();

      float8 d = zf;
#pragma unroll
      for (int ks = 0; ks < 2; ++ks)
        d = wmma16(fragA(t2o + mt * 16 * 64, 64, ks * 32, lane),
                   fragB(s_w + WT_T3, 64, 0, ks * 32, lane), d);
      storeD(tout + mt * 16 * 16, 16, d, s_b + B_T3, 0, false, lane);
      __builtin_amdgcn_wave_barrier();
    }
  }

  // =====================================================================
  // Stage 5: final MLP 32->16->16->2 + softmax
  //          input row s = concat(team0_out[s], team1_out[s])
  // =====================================================================
  {
    _Float16* f1o = scr;        // [16][16]
    _Float16* f2o = scr + 256;  // [16][16]
    const int r = lane & 15, kb = (lane >> 4) * 8;
    half16 A = fragA2(tout + r * 16 + kb,          // K 0..15  : team 0
                      tout + (16 + r) * 16 + kb);  // K 16..31 : team 1
    float8 d = wmma16(A, fragB(s_w + WT_F1, 32, 0, 0, lane), zf);
    storeD(f1o, 16, d, s_b + B_F1, 0, true, lane);
    __builtin_amdgcn_wave_barrier();
    d = wmma16(fragA_k16(f1o, 16, lane), fragB(s_w + WT_F2, 32, 0, 0, lane), zf);
    storeD(f2o, 16, d, s_b + B_F2, 0, true, lane);
    __builtin_amdgcn_wave_barrier();
    d = wmma16(fragA_k16(f2o, 16, lane), fragB(s_w + WT_F3, 32, 0, 0, lane), zf);

    int c = lane & 15, r0 = (lane >> 4) * 8;
    float bv = s_b[B_F3 + c];
#pragma unroll
    for (int e = 0; e < 8; ++e) {
      float lg    = d[e] + bv;
      float other = __shfl_xor(lg, 1, 32);   // cols 0<->1 carry the two logits
      float m  = fmaxf(lg, other);
      float e0 = __expf(lg - m);
      float e1 = __expf(other - m);
      if (c < 2)
        out[(size_t)(sampleBase + r0 + e) * 2 + c] = e0 / (e0 + e1);
    }
  }
}

// ---------------------------------------------------------------------------
extern "C" void kernel_launch(void* const* d_in, const int* in_sizes, int n_in,
                              void* d_out, int out_size, void* d_ws, size_t ws_size,
                              hipStream_t stream) {
  (void)n_in; (void)out_size; (void)d_ws; (void)ws_size;
  const float* x   = (const float*)d_in[0];
  const float* hw1 = (const float*)d_in[1];
  const float* hb1 = (const float*)d_in[2];
  const float* hw2 = (const float*)d_in[3];
  const float* hb2 = (const float*)d_in[4];
  const float* hw3 = (const float*)d_in[5];
  const float* hb3 = (const float*)d_in[6];
  const float* cw1 = (const float*)d_in[7];
  const float* cb1 = (const float*)d_in[8];
  const float* cw2 = (const float*)d_in[9];
  const float* cb2 = (const float*)d_in[10];
  const float* cw3 = (const float*)d_in[11];
  const float* cb3 = (const float*)d_in[12];
  const float* tw1 = (const float*)d_in[13];
  const float* tb1 = (const float*)d_in[14];
  const float* tw2 = (const float*)d_in[15];
  const float* tb2 = (const float*)d_in[16];
  const float* tw3 = (const float*)d_in[17];
  const float* tb3 = (const float*)d_in[18];
  const float* fw1 = (const float*)d_in[19];
  const float* fb1 = (const float*)d_in[20];
  const float* fw2 = (const float*)d_in[21];
  const float* fb2 = (const float*)d_in[22];
  const float* fw3 = (const float*)d_in[23];
  const float* fb3 = (const float*)d_in[24];

  const int nSamples = in_sizes[0] / 1400;       // B (x is [B,10,140])
  const int blocks   = (nSamples + 63) / 64;     // 4 waves/block * 16 samples/wave

  draft_fused_kernel<<<blocks, 128, 0, stream>>>(
      x, hw1, hb1, hw2, hb2, hw3, hb3,
      cw1, cb1, cw2, cb2, cw3, cb3,
      tw1, tb1, tw2, tb2, tw3, tb3,
      fw1, fb1, fw2, fb2, fw3, fb3,
      (float*)d_out, nSamples);
}
